// FixedFreqModel_77068893160256
// MI455X (gfx1250) — matrix operand order repaired and checked by reading, not compiled, not run on hardware
//
#include <hip/hip_runtime.h>
#include <hip/hip_bf16.h>
#include <stdint.h>

typedef __bf16 bf16_t;
typedef __attribute__((ext_vector_type(16))) __bf16 v16bf;
typedef __attribute__((ext_vector_type(8)))  float  v8f;

#define H 64
#define MEMN 256
#define BATCH 512
#define SEQLEN 2048
#define VOCAB 64
#define RF 16
#define CHUNKS (SEQLEN / RF)

// B-fragment tile table in workspace (each tile = 256 dwords = 32 lanes x 8 dwords)
#define WHH_T   0     // 24 tiles  (192x64  -> nt 0..11, kt 0..1)
#define WIX_T   24    // 24 tiles  (w_ih cols 0..63)
#define WIR_T   48    // 24 tiles  (w_ih cols 64..127)
#define WQ_T    72    // 8 tiles
#define WHEAD_T 80    // 8 tiles
#define WKEY_T  88    // 8 tiles
#define WVAL_T  96    // 8 tiles
#define NFRAG_TILES 104
#define EMB_OFF (NFRAG_TILES * 256)   // 26624 dwords: 64x64 bf16 embed table (packed pairs)
#define KV_OFF  (EMB_OFF + 2048)      // 28672 dwords: then k_bf, v_bf (bf16)

__device__ __forceinline__ uint16_t f2bfbits(float f) {
    union { float f; uint32_t u; } c; c.f = f;
    uint32_t u = c.u;
    u += 0x7FFFu + ((u >> 16) & 1u);   // round-to-nearest-even
    return (uint16_t)(u >> 16);
}
__device__ __forceinline__ bf16_t f2bf(float f) {
    union { uint16_t u; bf16_t b; } c; c.u = f2bfbits(f); return c.b;
}
__device__ __forceinline__ float bflo(uint32_t u) {
    union { uint32_t u; float f; } c; c.u = u << 16; return c.f;
}
__device__ __forceinline__ float bfhi(uint32_t u) {
    union { uint32_t u; float f; } c; c.u = u & 0xFFFF0000u; return c.f;
}

// A fragment (16x32 bf16) from a row-major [16][64] bf16 LDS tile.
// lanes 0-15: row=lane, K = kt*32 + {0..7, 16..23}; lanes 16-31: row=lane-16, +8.
__device__ __forceinline__ v16bf a_frag_lds(const bf16_t* tile, int kt) {
    int lane = threadIdx.x & 31;
    int row  = lane & 15;
    int k0   = kt * 32 + (lane >> 4) * 8;
    const uint32_t* p0 = (const uint32_t*)(tile + row * 64 + k0);
    const uint32_t* p1 = (const uint32_t*)(tile + row * 64 + k0 + 16);
    union { uint32_t u[8]; v16bf v; } f;
#pragma unroll
    for (int i = 0; i < 4; ++i) { f.u[i] = p0[i]; f.u[4 + i] = p1[i]; }
    return f.v;
}

// B fragment (32x16 bf16) from pre-swizzled workspace: lane-contiguous 8 dwords.
__device__ __forceinline__ v16bf b_frag_ws(const uint32_t* ws, int tile) {
    int lane = threadIdx.x & 31;
    const uint32_t* p = ws + (size_t)tile * 256 + lane * 8;
    union { uint32_t u[8]; v16bf v; } f;
#pragma unroll
    for (int i = 0; i < 8; ++i) f.u[i] = p[i];
    return f.v;
}

__device__ __forceinline__ v8f wmma_bf16(v16bf a, v16bf b, v8f c) {
    return __builtin_amdgcn_wmma_f32_16x16x32_bf16(false, a, false, b, (short)0, c, false, false);
}

// C/D accumulator preloaded with a per-column bias (same bias for all 8 rows a lane holds)
__device__ __forceinline__ v8f bcast_col(const float* bias, int col0) {
    int lane = threadIdx.x & 31;
    float x = bias[col0 + (lane & 15)];
    v8f c;
#pragma unroll
    for (int i = 0; i < 8; ++i) c[i] = x;
    return c;
}

// ---------------------------------------------------------------------------
// Kernel 1: swizzle all weights into WMMA B-fragment order (bf16) + embed table
// ---------------------------------------------------------------------------
__global__ __launch_bounds__(256)
void prep_kernel(const float* __restrict__ w_ih, const float* __restrict__ w_hh,
                 const float* __restrict__ qw,  const float* __restrict__ hw,
                 const float* __restrict__ kw,  const float* __restrict__ vw,
                 const float* __restrict__ emb, uint32_t* __restrict__ ws) {
    int id = blockIdx.x * 256 + threadIdx.x;
    if (id < NFRAG_TILES * 256) {
        int tile = id >> 8, idx = id & 255;
        int lane = idx >> 3, v = idx & 7;
        const float* W; int stride, col0, ltile;
        if (tile < 24)      { W = w_hh; stride = 64;  col0 = 0;  ltile = tile; }
        else if (tile < 48) { W = w_ih; stride = 128; col0 = 0;  ltile = tile - 24; }
        else if (tile < 72) { W = w_ih; stride = 128; col0 = 64; ltile = tile - 48; }
        else if (tile < 80) { W = qw;   stride = 64;  col0 = 0;  ltile = tile - 72; }
        else if (tile < 88) { W = hw;   stride = 64;  col0 = 0;  ltile = tile - 80; }
        else if (tile < 96) { W = kw;   stride = 64;  col0 = 0;  ltile = tile - 88; }
        else                { W = vw;   stride = 64;  col0 = 0;  ltile = tile - 96; }
        int nt = ltile >> 1, kt = ltile & 1;
        int n = nt * 16 + (lane & 15);              // B column = lane%16
        int k = kt * 32 + (lane >> 4) * 8 + (v & 3) * 2 + (v >> 2) * 16; // packed K pair per vgpr
        float lo = W[n * stride + col0 + k];
        float hi = W[n * stride + col0 + k + 1];
        ws[(size_t)tile * 256 + idx] = (uint32_t)f2bfbits(lo) | ((uint32_t)f2bfbits(hi) << 16);
    } else if (id < KV_OFF) {
        int e = id - EMB_OFF;
        int row = e >> 5, pr = e & 31;
        float lo = emb[row * 64 + pr * 2];
        float hi = emb[row * 64 + pr * 2 + 1];
        ws[EMB_OFF + e] = (uint32_t)f2bfbits(lo) | ((uint32_t)f2bfbits(hi) << 16);
    }
}

// ---------------------------------------------------------------------------
// Kernel 2: k = memory @ Wk^T + bk ; v = memory @ Wv^T + bv   (bf16 out, WMMA)
// one wave per 16-row tile; all B fragments hoisted so the 16 wmmas issue
// back-to-back after a single load wait.
// ---------------------------------------------------------------------------
__global__ __launch_bounds__(128, 1)
void kv_kernel(const float* __restrict__ memory,
               const float* __restrict__ key_b, const float* __restrict__ val_b,
               const uint32_t* __restrict__ ws,
               uint16_t* __restrict__ kbf, uint16_t* __restrict__ vbf) {
    __shared__ __attribute__((aligned(16))) bf16_t mtile[4][16 * 64];
    const int tid = threadIdx.x, wave = tid >> 5, lane = tid & 31;
    const size_t m0 = ((size_t)blockIdx.x * 4 + wave) * 16;

    // hoist all weight fragments (issued as one load clause)
    v16bf bk[4][2], bv[4][2];
#pragma unroll
    for (int nt = 0; nt < 4; ++nt)
#pragma unroll
        for (int kt = 0; kt < 2; ++kt) {
            bk[nt][kt] = b_frag_ws(ws, WKEY_T + nt * 2 + kt);
            bv[nt][kt] = b_frag_ws(ws, WVAL_T + nt * 2 + kt);
        }

    for (int i = lane; i < 16 * 64; i += 32)
        mtile[wave][i] = f2bf(memory[m0 * 64 + i]);
    __syncthreads();

    v16bf a0 = a_frag_lds(mtile[wave], 0);
    v16bf a1 = a_frag_lds(mtile[wave], 1);

    v8f ka[4], va[4];
#pragma unroll
    for (int nt = 0; nt < 4; ++nt) {
        ka[nt] = bcast_col(key_b, nt * 16);
        va[nt] = bcast_col(val_b, nt * 16);
    }
#pragma unroll
    for (int nt = 0; nt < 4; ++nt) {
        ka[nt] = wmma_bf16(a0, bk[nt][0], ka[nt]);
        ka[nt] = wmma_bf16(a1, bk[nt][1], ka[nt]);
        va[nt] = wmma_bf16(a0, bv[nt][0], va[nt]);
        va[nt] = wmma_bf16(a1, bv[nt][1], va[nt]);
    }

    const int rb8 = (lane >> 4) * 8;
#pragma unroll
    for (int nt = 0; nt < 4; ++nt) {
        int col = nt * 16 + (lane & 15);
#pragma unroll
        for (int r = 0; r < 8; ++r) {
            size_t row = m0 + rb8 + r;
            kbf[row * 64 + col] = f2bfbits(ka[nt][r]);
            vbf[row * 64 + col] = f2bfbits(va[nt][r]);
        }
    }
}

// ---------------------------------------------------------------------------
// Kernel 3: fused recurrent model. 1 block = 16 batch rows, 4 waves.
// Wave w owns gate tiles nt {w, 4+w, 8+w} => hidden cols [16w,16w+16):
// elementwise GRU update is wave-local; h round-trips LDS as bf16.
// launch_bounds(128,1): allow a large VGPR budget so the 12 persistent
// weight fragments stay resident across the 2048-step recurrence.
// ---------------------------------------------------------------------------
__global__ __launch_bounds__(128, 1)
void rnn_main_kernel(const int* __restrict__ seq,
                     const float* __restrict__ b_ih, const float* __restrict__ b_hh,
                     const float* __restrict__ query_b, const float* __restrict__ head_b,
                     const uint32_t* __restrict__ ws,
                     const uint16_t* __restrict__ kbf, const uint16_t* __restrict__ vbf,
                     float* __restrict__ out) {
    __shared__ __attribute__((aligned(16))) uint32_t emb_table[VOCAB * 32]; // packed bf16 pairs
    __shared__ __attribute__((aligned(16))) uint32_t emb_tile_u[16 * 32];
    __shared__ __attribute__((aligned(16))) bf16_t  h_bf[16 * 64];
    __shared__ __attribute__((aligned(16))) bf16_t  ret_bf[16 * 64];
    __shared__ __attribute__((aligned(16))) float   q_lds[16 * 64];
    __shared__ float sc[16 * MEMN];
    __shared__ float red[16 * 8];
    __shared__ float rowmax[16];
    __shared__ float rowsum[16];
    __shared__ int   tok[16 * 16];

    const int tid = threadIdx.x, wave = tid >> 5, lane = tid & 31;
    const int b0 = blockIdx.x * 16;
    const int colw = wave * 16 + (lane & 15);
    const int rb8  = (lane >> 4) * 8;
    bf16_t* emb_tile = (bf16_t*)emb_tile_u;

    for (int i = tid; i < VOCAB * 32; i += 128) emb_table[i] = ws[EMB_OFF + i];
    for (int i = tid; i < 16 * 64;    i += 128) h_bf[i] = f2bf(0.f);

    // persistent per-wave weight fragments + bias tiles
    v16bf wix[3][2], whh[3][2];
#pragma unroll
    for (int g = 0; g < 3; ++g)
#pragma unroll
        for (int kt = 0; kt < 2; ++kt) {
            wix[g][kt] = b_frag_ws(ws, WIX_T + (g * 4 + wave) * 2 + kt);
            whh[g][kt] = b_frag_ws(ws, WHH_T + (g * 4 + wave) * 2 + kt);
        }
    v16bf wq0 = b_frag_ws(ws, WQ_T + wave * 2 + 0);
    v16bf wq1 = b_frag_ws(ws, WQ_T + wave * 2 + 1);
    v8f bihT[3], bhhT[3];
#pragma unroll
    for (int g = 0; g < 3; ++g) {
        bihT[g] = bcast_col(b_ih, g * 64 + wave * 16);
        bhhT[g] = bcast_col(b_hh, g * 64 + wave * 16);
    }
    v8f qbias = bcast_col(query_b, wave * 16);
    v8f hD;
#pragma unroll
    for (int i = 0; i < 8; ++i) hD[i] = 0.f;
    __syncthreads();

    for (int c = 0; c < CHUNKS; ++c) {
        // tokens for this chunk
        for (int e = tid; e < 256; e += 128)
            tok[e] = seq[(size_t)(b0 + (e >> 4)) * SEQLEN + c * RF + (e & 15)];

        // q = h @ Wq^T + bq   (each wave -> 16 cols)
        {
            v16bf ah0 = a_frag_lds(h_bf, 0), ah1 = a_frag_lds(h_bf, 1);
            v8f q = qbias;
            q = wmma_bf16(ah0, wq0, q);
            q = wmma_bf16(ah1, wq1, q);
#pragma unroll
            for (int r = 0; r < 8; ++r) q_lds[(rb8 + r) * 64 + colw] = q[r];
        }
        __syncthreads();

        // scores = (q . k) * scale
        {
            const int br = tid >> 3, sg = tid & 7;
            const float* qr = q_lds + br * 64;
            float pmax = -3.4e38f;
            for (int mm = 0; mm < 32; ++mm) {
                int m = sg * 32 + mm;
                const uint32_t* kp = (const uint32_t*)(kbf + ((size_t)(b0 + br) * MEMN + m) * 64);
                __builtin_prefetch(kp + 32, 0, 0);   // next k row (128B) -> global_prefetch
                float s = 0.f;
#pragma unroll
                for (int j = 0; j < 32; ++j) {
                    uint32_t u = kp[j];
                    s += qr[2 * j] * bflo(u) + qr[2 * j + 1] * bfhi(u);
                }
                s *= 0.125f;
                sc[br * MEMN + m] = s;
                pmax = fmaxf(pmax, s);
            }
            red[br * 8 + sg] = pmax;
        }
        __syncthreads();
        if (tid < 16) {
            float mx = red[tid * 8];
#pragma unroll
            for (int j = 1; j < 8; ++j) mx = fmaxf(mx, red[tid * 8 + j]);
            rowmax[tid] = mx;
        }
        __syncthreads();
        {
            const int br = tid >> 3, sg = tid & 7;
            float mx = rowmax[br], ps = 0.f;
            for (int mm = 0; mm < 32; ++mm) {
                int m = sg * 32 + mm;
                float e = __expf(sc[br * MEMN + m] - mx);
                sc[br * MEMN + m] = e;
                ps += e;
            }
            red[br * 8 + sg] = ps;
        }
        __syncthreads();
        if (tid < 16) {
            float s = 0.f;
#pragma unroll
            for (int j = 0; j < 8; ++j) s += red[tid * 8 + j];
            rowsum[tid] = s;
        }
        __syncthreads();
        // retrieved = softmax(scores) @ v
        {
            const int br = tid >> 3, h0 = (tid & 7) * 8;
            float inv = 1.f / rowsum[br];
            float acc[8];
#pragma unroll
            for (int i = 0; i < 8; ++i) acc[i] = 0.f;
            for (int m = 0; m < MEMN; ++m) {
                float wgt = sc[br * MEMN + m];
                const uint32_t* vp = (const uint32_t*)(vbf + ((size_t)(b0 + br) * MEMN + m) * 64 + h0);
#pragma unroll
                for (int j = 0; j < 4; ++j) {
                    uint32_t u = vp[j];
                    acc[2 * j]     += wgt * bflo(u);
                    acc[2 * j + 1] += wgt * bfhi(u);
                }
            }
#pragma unroll
            for (int i = 0; i < 8; ++i) ret_bf[br * 64 + h0 + i] = f2bf(acc[i] * inv);
        }
        __syncthreads();

        // giBase = ret @ Wir^T + b_ih   (constant over the chunk); hoist frags
        v8f gib[3];
        {
            v16bf wir[3][2];
#pragma unroll
            for (int g = 0; g < 3; ++g)
#pragma unroll
                for (int kt = 0; kt < 2; ++kt)
                    wir[g][kt] = b_frag_ws(ws, WIR_T + (g * 4 + wave) * 2 + kt);
            v16bf ar0 = a_frag_lds(ret_bf, 0), ar1 = a_frag_lds(ret_bf, 1);
#pragma unroll
            for (int g = 0; g < 3; ++g) {
                v8f t = bihT[g];
                t = wmma_bf16(ar0, wir[g][0], t);
                t = wmma_bf16(ar1, wir[g][1], t);
                gib[g] = t;
            }
        }

        // 16 sequential GRU steps
        for (int t = 0; t < RF; ++t) {
            {   // gather embedding tile (16 tokens x 64 dims, bf16)
                int r = tid >> 3, sg = tid & 7;
                int tk = tok[r * 16 + t];
#pragma unroll
                for (int i = 0; i < 4; ++i)
                    emb_tile_u[r * 32 + sg * 4 + i] = emb_table[tk * 32 + sg * 4 + i];
            }
            __syncthreads();
            v16bf ae0 = a_frag_lds(emb_tile, 0), ae1 = a_frag_lds(emb_tile, 1);
            v16bf ah0 = a_frag_lds(h_bf, 0),     ah1 = a_frag_lds(h_bf, 1);
            v8f gi[3], gh[3];
#pragma unroll
            for (int g = 0; g < 3; ++g) {
                v8f t1 = wmma_bf16(ae0, wix[g][0], gib[g]);
                gi[g]  = wmma_bf16(ae1, wix[g][1], t1);
                v8f t2 = wmma_bf16(ah0, whh[g][0], bhhT[g]);
                gh[g]  = wmma_bf16(ah1, whh[g][1], t2);
            }
#pragma unroll
            for (int i = 0; i < 8; ++i) {
                float rr = 1.f / (1.f + __expf(-(gi[0][i] + gh[0][i])));
                float zz = 1.f / (1.f + __expf(-(gi[1][i] + gh[1][i])));
                float ax = gi[2][i] + rr * gh[2][i];
                float nn = 2.f / (1.f + __expf(-2.f * ax)) - 1.f;
                hD[i] = (1.f - zz) * nn + zz * hD[i];
            }
            __syncthreads();
#pragma unroll
            for (int r = 0; r < 8; ++r) h_bf[(rb8 + r) * 64 + colw] = f2bf(hD[r]);
            __syncthreads();
        }
    }

    // logits = h @ Whead^T + b_head
    {
        v16bf ah0 = a_frag_lds(h_bf, 0), ah1 = a_frag_lds(h_bf, 1);
        v8f o = bcast_col(head_b, wave * 16);
        o = wmma_bf16(ah0, b_frag_ws(ws, WHEAD_T + wave * 2 + 0), o);
        o = wmma_bf16(ah1, b_frag_ws(ws, WHEAD_T + wave * 2 + 1), o);
#pragma unroll
        for (int r = 0; r < 8; ++r)
            out[(size_t)(b0 + rb8 + r) * VOCAB + colw] = o[r];
    }
}

extern "C" void kernel_launch(void* const* d_in, const int* in_sizes, int n_in,
                              void* d_out, int out_size, void* d_ws, size_t ws_size,
                              hipStream_t stream) {
    (void)in_sizes; (void)n_in; (void)out_size; (void)ws_size;
    const int*   seq      = (const int*)  d_in[0];
    const float* memory   = (const float*)d_in[1];
    const float* embed_w  = (const float*)d_in[2];
    const float* gru_w_ih = (const float*)d_in[3];
    const float* gru_w_hh = (const float*)d_in[4];
    const float* gru_b_ih = (const float*)d_in[5];
    const float* gru_b_hh = (const float*)d_in[6];
    const float* key_w    = (const float*)d_in[7];
    const float* key_b    = (const float*)d_in[8];
    const float* val_w    = (const float*)d_in[9];
    const float* val_b    = (const float*)d_in[10];
    const float* query_w  = (const float*)d_in[11];
    const float* query_b  = (const float*)d_in[12];
    const float* head_w   = (const float*)d_in[13];
    const float* head_b   = (const float*)d_in[14];
    float* out = (float*)d_out;

    uint32_t* ws  = (uint32_t*)d_ws;
    uint16_t* kbf = (uint16_t*)(ws + KV_OFF);
    uint16_t* vbf = kbf + (size_t)BATCH * MEMN * H;

    prep_kernel<<<112, 256, 0, stream>>>(gru_w_ih, gru_w_hh, query_w, head_w,
                                         key_w, val_w, embed_w, ws);
    kv_kernel<<<(BATCH * MEMN) / 64, 128, 0, stream>>>(memory, key_b, val_b, ws, kbf, vbf);
    rnn_main_kernel<<<BATCH / 16, 128, 0, stream>>>(seq, gru_b_ih, gru_b_hh,
                                                    query_b, head_b, ws, kbf, vbf, out);
}
